// GlobalAttentionBlock_35390530519684
// MI455X (gfx1250) — matrix operand order repaired and checked by reading, not compiled, not run on hardware
//
#include <hip/hip_runtime.h>
#include <stdint.h>

typedef __bf16 bf16_t;
typedef __attribute__((ext_vector_type(16))) __bf16    v16bf;
typedef __attribute__((ext_vector_type(8)))  __bf16    v8bf;
typedef __attribute__((ext_vector_type(8)))  float     v8f;
typedef __attribute__((ext_vector_type(4)))  uint32_t  u32x4;

#define CH     256
#define NPIX   4096
#define NHEADS 8
#define HD     32
#define NGRP   32
#define GSIZE  (CH / NGRP)
#define GN_EPS 1e-5f

// ---------------------------------------------------------------- helpers

// native f32->bf16 (RNE) -> v_cvt_pk_bf16_f32 when paired
__device__ __forceinline__ bf16_t f2bf(float f) { return (bf16_t)f; }
__device__ __forceinline__ uint32_t bfbits(float f) {
    union { bf16_t b; uint16_t s; } r; r.b = (bf16_t)f;
    return (uint32_t)r.s;
}

union U4 { u32x4 u; bf16_t h[8]; };
union F4 { u32x4 u; float  f[4]; };

// A fragment (16 rows x 32 K, 16-bit) from row-major tile (LDS or global), row stride ldk.
// lane&15 = row ; half: elems e -> K=kbase+half*16+e (contiguous). Two b128 loads.
__device__ __forceinline__ v16bf a_frag(const bf16_t* p0, int ldk, int kbase, int lane) {
    const bf16_t* p = p0 + (lane & 15) * ldk + kbase + ((lane >> 4) * 16);
    v8bf lo = *(const v8bf*)p;
    v8bf hi = *(const v8bf*)(p + 8);
    return __builtin_shufflevector(lo, hi, 0,1,2,3,4,5,6,7,8,9,10,11,12,13,14,15);
}

// B fragment (32 K x 16 N) from transposed-layout tile [N][K] (row stride ldk):
// lane&15 selects col ; elems e -> K = kbase + half*16 + e : 16 contiguous bf16.
__device__ __forceinline__ v16bf b_frag(const bf16_t* p0, int ldk, int col, int kbase, int lane) {
    return *(const v16bf*)(p0 + (size_t)col * ldk + kbase + ((lane >> 4) * 16));
}

__device__ __forceinline__ v8f wmma_bf16(v16bf a, v16bf b, v8f c) {
    return __builtin_amdgcn_wmma_f32_16x16x32_bf16(false, a, false, b, (short)0, c, false, false);
}

// per-lane async global->LDS 16B copy (ASYNCcnt-tracked); GV mode, 64-bit vaddr
__device__ __forceinline__ void async_b128(const void* lds_dst, const void* gsrc) {
    uint32_t lds = (uint32_t)(uintptr_t)lds_dst;        // aperture low bits == LDS byte addr
    uint64_t ga  = (uint64_t)(uintptr_t)gsrc;
    asm volatile("global_load_async_to_lds_b128 %0, %1, off"
                 :: "v"(lds), "v"(ga) : "memory");
}
__device__ __forceinline__ void wait_async0() {
    asm volatile("s_wait_asynccnt 0" ::: "memory");
}

// ---------------------------------------------------------------- kernel 1: GroupNorm -> bf16 pixel-major [b][n][c]
__global__ void k_groupnorm(const float* __restrict__ x,
                            const float* __restrict__ gn_w,
                            const float* __restrict__ gn_b,
                            bf16_t* __restrict__ xn,
                            float* __restrict__ gchan) {
    int bg = blockIdx.x;
    int b = bg / NGRP, g = bg % NGRP;
    int c0 = g * GSIZE;
    const float* xp = x + ((size_t)b * CH + c0) * NPIX;
    int t = threadIdx.x;

    float s = 0.f, ss = 0.f;
    for (int i = t; i < GSIZE * NPIX; i += 256) {
        float v = xp[i];
        s += v; ss += v * v;
    }
    __shared__ float red[256];
    red[t] = s; __syncthreads();
    for (int o = 128; o > 0; o >>= 1) { if (t < o) red[t] += red[t + o]; __syncthreads(); }
    float mean = red[0] * (1.0f / (GSIZE * NPIX));
    __syncthreads();
    red[t] = ss; __syncthreads();
    for (int o = 128; o > 0; o >>= 1) { if (t < o) red[t] += red[t + o]; __syncthreads(); }
    float var = red[0] * (1.0f / (GSIZE * NPIX)) - mean * mean;
    float inv = rsqrtf(var + GN_EPS);

    __shared__ float csum[GSIZE];
    if (t < GSIZE) csum[t] = 0.f;
    __syncthreads();

    float gw[GSIZE], gb[GSIZE], csl[GSIZE];
#pragma unroll
    for (int ch = 0; ch < GSIZE; ++ch) { gw[ch] = gn_w[c0 + ch]; gb[ch] = gn_b[c0 + ch]; csl[ch] = 0.f; }

    for (int j = t; j < NPIX; j += 256) {
        U4 pk;
#pragma unroll
        for (int ch = 0; ch < GSIZE; ++ch) {
            float v = (xp[ch * NPIX + j] - mean) * inv * gw[ch] + gb[ch];
            pk.h[ch] = f2bf(v);
            csl[ch] += v;
        }
        *(u32x4*)(xn + ((size_t)b * NPIX + j) * CH + c0) = pk.u;
    }
#pragma unroll
    for (int ch = 0; ch < GSIZE; ++ch) {
        float loc = csl[ch];
#pragma unroll
        for (int o = 16; o > 0; o >>= 1) loc += __shfl_xor(loc, o, 32);
        if ((t & 31) == 0) atomicAdd(&csum[ch], loc);
    }
    __syncthreads();
    if (t < GSIZE) gchan[b * CH + c0 + t] = csum[t];
}

// ---------------------------------------------------------------- kernel 2: gfeat
__global__ void k_gfeat(const float* __restrict__ gchan,
                        const float* __restrict__ gproj_w,
                        const float* __restrict__ gproj_b,
                        float* __restrict__ gfeat) {
    int b = blockIdx.x;
    int o = threadIdx.x;
    float s = 0.f;
    for (int c = 0; c < CH; ++c) s += gchan[b * CH + c] * gproj_w[o * CH + c];
    gfeat[b * CH + o] = s * (1.0f / NPIX) + gproj_b[o];
}

// ---------------------------------------------------------------- kernel 3: QKV GEMM (transposed form D^T = X^T W^T)
// grid (NPIX/32, 768/256, B) ; block 256 = 8 waves as 2(j) x 4(o)
__global__ void k_qkv_gemm(const bf16_t* __restrict__ xn,   // [b][n][c]
                           const float* __restrict__ qkv_w,
                           const float* __restrict__ qkv_b,
                           const float* __restrict__ gfeat,
                           bf16_t* __restrict__ qb,         // [b][h][n][d]
                           bf16_t* __restrict__ kb,         // [b][h][n][d]
                           bf16_t* __restrict__ vb) {       // [b][h][d][n]
    int j0 = blockIdx.x * 32;
    int o0 = blockIdx.y * 256;
    int b  = blockIdx.z;
    int t = threadIdx.x, lane = t & 31, wave = t >> 5;
    int wmj = (wave >> 2) * 16;
    int wno = (wave & 3) * 64;

    __shared__ __align__(32) bf16_t Ws[256 * 32];   // W^T in natural [o][c] layout, fp32->bf16

    const bf16_t* Arow = xn + ((size_t)b * NPIX + j0 + wmj) * CH;

    v8f acc[4] = {{}, {}, {}, {}};
    for (int kc = 0; kc < CH; kc += 32) {
        __syncthreads();
#pragma unroll
        for (int r = 0; r < 4; ++r) {
            int g = t + r * 256;
            int o = g >> 2, cseg = (g & 3) * 8;
            const float* wp = qkv_w + (size_t)(o0 + o) * CH + kc + cseg;
            U4 pk;
#pragma unroll
            for (int e = 0; e < 8; ++e) pk.h[e] = f2bf(wp[e]);
            *(u32x4*)(Ws + o * 32 + cseg) = pk.u;
        }
        __syncthreads();

        v16bf af = a_frag(Arow, CH, kc, lane);
#pragma unroll
        for (int s = 0; s < 4; ++s) {
            v16bf bf = b_frag(Ws, 32, wno + s * 16 + (lane & 15), 0, lane);
            acc[s] = wmma_bf16(af, bf, acc[s]);
        }
    }

    int jrow = j0 + wmj + (lane >> 4) * 8;
#pragma unroll
    for (int s = 0; s < 4; ++s) {
        int og = o0 + wno + s * 16 + (lane & 15);
        float bias = qkv_b[og];
        if (og < CH) {                                    // Q [b,h,n,d]
            int hh = og >> 5, d = og & 31;
            bf16_t* qp = qb + ((size_t)(b * NHEADS + hh) * NPIX) * HD + d;
#pragma unroll
            for (int r = 0; r < 8; ++r)
                qp[(size_t)(jrow + r) * HD] = f2bf(acc[s][r] + bias);
        } else if (og < 2 * CH) {                         // K [b,h,n,d] + 0.1*gfeat
            int oc = og - CH;
            int hh = oc >> 5, d = oc & 31;
            float gf = 0.1f * gfeat[b * CH + oc];
            bf16_t* kp = kb + ((size_t)(b * NHEADS + hh) * NPIX) * HD + d;
#pragma unroll
            for (int r = 0; r < 8; ++r)
                kp[(size_t)(jrow + r) * HD] = f2bf(acc[s][r] + bias + gf);
        } else {                                          // V [b,h,d,n]: packed b128 store
            int ov = og - 2 * CH;
            int hh = ov >> 5, d = ov & 31;
            U4 pk;
#pragma unroll
            for (int i = 0; i < 8; ++i) pk.h[i] = f2bf(acc[s][i] + bias);
            *(u32x4*)(vb + ((size_t)(b * NHEADS + hh) * HD + d) * NPIX + jrow) = pk.u;
        }
    }
}

// ---------------------------------------------------------------- kernel 4: flash attention, S^T = K Q^T, O^T = V^T P^T
// grid (NPIX/128, B*NHEADS) ; block 256 = 8 waves x 16 queries (1 query per lane pair)
__global__ void k_attn(const bf16_t* __restrict__ qb,
                       const bf16_t* __restrict__ kb,
                       const bf16_t* __restrict__ vb,
                       bf16_t* __restrict__ ao) {          // [b][n][c]
    int bh = blockIdx.y;
    int b = bh / NHEADS, h = bh % NHEADS;
    int q0 = blockIdx.x * 128;
    int t = threadIdx.x, lane = t & 31, wave = t >> 5;
    const float scale = 0.17677669529663687f;

    const bf16_t* Q = qb + (size_t)bh * NPIX * HD;         // [n][d]
    const bf16_t* K = kb + (size_t)bh * NPIX * HD;         // [n][d]
    const bf16_t* V = vb + (size_t)bh * HD * NPIX;         // [d][n]

    int qidx = q0 + wave * 16 + (lane & 15);
    v16bf qf = *(const v16bf*)(Q + (size_t)qidx * HD + ((lane >> 4) * 16));

    __shared__ __align__(32) bf16_t Ks[64 * 32];    // [key][d]
    __shared__ __align__(32) bf16_t Vs[32 * 64];    // [d][key]

    // per-thread staging addresses (loop-invariant parts)
    int vrow = t >> 3, vseg = (t & 7) * 8;
    bf16_t* ksDst = Ks + t * 8;
    bf16_t* vsDst = Vs + vrow * 64 + vseg;

    float m_run = -1e30f, l_run = 0.f;
    v8f o_acc0 = {}, o_acc1 = {};
    bool h1 = (lane >= 16);

    for (int kc = 0; kc < NPIX; kc += 64) {
        __syncthreads();
        // async DMA: global -> LDS, 16B per lane, ASYNCcnt-tracked
        async_b128(ksDst, K + (size_t)kc * HD + t * 8);
        async_b128(vsDst, V + (size_t)vrow * NPIX + kc + vseg);
        if (kc + 64 < NPIX) {
            __builtin_prefetch(K + (size_t)(kc + 64) * HD, 0, 1);
            __builtin_prefetch(V + (size_t)(kc + 64), 0, 1);
        }
        wait_async0();
        __syncthreads();

        // S^T (64 key x 16 q): 4 M-tiles; lane owns one query column
        v8f sv[4];
#pragma unroll
        for (int mt = 0; mt < 4; ++mt) {
            v16bf kf = a_frag(Ks + mt * 16 * 32, 32, 0, lane);
            v8f z = {};
            sv[mt] = wmma_bf16(kf, qf, z);
        }

        float mx = -1e30f;
#pragma unroll
        for (int mt = 0; mt < 4; ++mt)
#pragma unroll
            for (int r = 0; r < 8; ++r) { sv[mt][r] *= scale; mx = fmaxf(mx, sv[mt][r]); }
        mx = fmaxf(mx, __shfl_xor(mx, 16, 32));
        float mnew = fmaxf(m_run, mx);
        float corr = __expf(m_run - mnew);
        m_run = mnew;

        float pv[4][8];
        float ps = 0.f;
#pragma unroll
        for (int mt = 0; mt < 4; ++mt)
#pragma unroll
            for (int r = 0; r < 8; ++r) { float p = __expf(sv[mt][r] - mnew); pv[mt][r] = p; ps += p; }
        ps += __shfl_xor(ps, 16, 32);
        l_run = l_run * corr + ps;
#pragma unroll
        for (int r = 0; r < 8; ++r) { o_acc0[r] *= corr; o_acc1[r] *= corr; }

        // P^T B-fragments in registers: half-swap via shfl_xor(16) + select + pack
#pragma unroll
        for (int kstep = 0; kstep < 2; ++kstep) {
            const float* pA = pv[2 * kstep];
            const float* pB = pv[2 * kstep + 1];
            float lo[8], hi[8];
#pragma unroll
            for (int e = 0; e < 8; ++e) {
                float qAe = __shfl_xor(pA[e], 16, 32);
                float qBe = __shfl_xor(pB[e], 16, 32);
                lo[e] = h1 ? qBe : pA[e];
                hi[e] = h1 ? pB[e] : qAe;
            }
            union { v16bf v; uint32_t u[8]; } pf;
#pragma unroll
            for (int i = 0; i < 4; ++i) {
                pf.u[i]     = bfbits(lo[2 * i]) | (bfbits(lo[2 * i + 1]) << 16);
                pf.u[4 + i] = bfbits(hi[2 * i]) | (bfbits(hi[2 * i + 1]) << 16);
            }
            v16bf vf0 = a_frag(Vs,           64, kstep * 32, lane);
            v16bf vf1 = a_frag(Vs + 16 * 64, 64, kstep * 32, lane);
            o_acc0 = wmma_bf16(vf0, pf.v, o_acc0);
            o_acc1 = wmma_bf16(vf1, pf.v, o_acc1);
        }
    }

    float invl = 1.0f / l_run;
    int half8 = (lane >> 4) * 8;
    bf16_t* aop = ao + ((size_t)b * NPIX + qidx) * CH + h * HD;
    U4 w0, w1;
#pragma unroll
    for (int i = 0; i < 8; ++i) { w0.h[i] = f2bf(o_acc0[i] * invl); w1.h[i] = f2bf(o_acc1[i] * invl); }
    *(u32x4*)(aop + half8)      = w0.u;
    *(u32x4*)(aop + 16 + half8) = w1.u;
}

// ---------------------------------------------------------------- kernel 5: projection (transposed form) + residual
// grid (NPIX/32, 1, B) ; block 256
__global__ void k_proj_gemm(const bf16_t* __restrict__ ao,  // [b][n][c]
                            const float* __restrict__ proj_w,
                            const float* __restrict__ proj_b,
                            const float* __restrict__ gfeat,
                            const float* __restrict__ x,
                            float* __restrict__ out) {
    int j0 = blockIdx.x * 32;
    int b  = blockIdx.z;
    int t = threadIdx.x, lane = t & 31, wave = t >> 5;
    int wmj = (wave >> 2) * 16;
    int wno = (wave & 3) * 64;

    __shared__ __align__(32) bf16_t Ws[256 * 32];

    const bf16_t* Arow = ao + ((size_t)b * NPIX + j0 + wmj) * CH;

    v8f acc[4] = {{}, {}, {}, {}};
    for (int kc = 0; kc < CH; kc += 32) {
        __syncthreads();
#pragma unroll
        for (int r = 0; r < 4; ++r) {
            int g = t + r * 256;
            int o = g >> 2, cseg = (g & 3) * 8;
            const float* wp = proj_w + (size_t)o * CH + kc + cseg;
            U4 pk;
#pragma unroll
            for (int e = 0; e < 8; ++e) pk.h[e] = f2bf(wp[e]);
            *(u32x4*)(Ws + o * 32 + cseg) = pk.u;
        }
        __syncthreads();

        v16bf af = a_frag(Arow, CH, kc, lane);
#pragma unroll
        for (int s = 0; s < 4; ++s) {
            v16bf bf = b_frag(Ws, 32, wno + s * 16 + (lane & 15), 0, lane);
            acc[s] = wmma_bf16(af, bf, acc[s]);
        }
    }

    int jrow = j0 + wmj + (lane >> 4) * 8;
#pragma unroll
    for (int s = 0; s < 4; ++s) {
        int og = wno + s * 16 + (lane & 15);
        float base = proj_b[og] + 0.1f * gfeat[b * CH + og];
        size_t idx = ((size_t)b * CH + og) * NPIX + jrow;
        F4 x0, x1, r0, r1;
        x0.u = *(const u32x4*)(x + idx);
        x1.u = *(const u32x4*)(x + idx + 4);
#pragma unroll
        for (int i = 0; i < 4; ++i) {
            r0.f[i] = acc[s][i]     + base + x0.f[i];
            r1.f[i] = acc[s][i + 4] + base + x1.f[i];
        }
        *(u32x4*)(out + idx)     = r0.u;
        *(u32x4*)(out + idx + 4) = r1.u;
    }
}

// ---------------------------------------------------------------- launch
extern "C" void kernel_launch(void* const* d_in, const int* in_sizes, int n_in,
                              void* d_out, int out_size, void* d_ws, size_t ws_size,
                              hipStream_t stream) {
    const float* x       = (const float*)d_in[0];
    const float* gn_w    = (const float*)d_in[1];
    const float* gn_b    = (const float*)d_in[2];
    const float* qkv_w   = (const float*)d_in[3];
    const float* qkv_b   = (const float*)d_in[4];
    const float* proj_w  = (const float*)d_in[5];
    const float* proj_b  = (const float*)d_in[6];
    const float* gproj_w = (const float*)d_in[7];
    const float* gproj_b = (const float*)d_in[8];

    char* ws = (char*)d_ws;
    const size_t MB = 1u << 20;
    bf16_t* xn  = (bf16_t*)(ws + 0 * MB);   // [b][n][c] bf16
    bf16_t* ao  = (bf16_t*)(ws + 0 * MB);   // aliased (xn consumed before attn writes)
    bf16_t* qb  = (bf16_t*)(ws + 4 * MB);   // [b][h][n][d]
    bf16_t* kb  = (bf16_t*)(ws + 8 * MB);   // [b][h][n][d]
    bf16_t* vb  = (bf16_t*)(ws + 12 * MB);  // [b][h][d][n]
    float*  gchan = (float*)(ws + 16 * MB);
    float*  gfeat = (float*)(ws + 16 * MB + 8192);
    float*  out = (float*)d_out;

    k_groupnorm<<<dim3(2 * NGRP), 256, 0, stream>>>(x, gn_w, gn_b, xn, gchan);
    k_gfeat<<<dim3(2), 256, 0, stream>>>(gchan, gproj_w, gproj_b, gfeat);
    k_qkv_gemm<<<dim3(NPIX / 32, 3, 2), 256, 0, stream>>>(
        xn, qkv_w, qkv_b, gfeat, qb, kb, vb);
    k_attn<<<dim3(NPIX / 128, 2 * NHEADS), 256, 0, stream>>>(qb, kb, vb, ao);
    k_proj_gemm<<<dim3(NPIX / 32, 1, 2), 256, 0, stream>>>(
        ao, proj_w, proj_b, gfeat, x, out);
}
